// VisionAttention_61254823575718
// MI455X (gfx1250) — compile-verified, matrix operand order checked by reading, not run on hardware
//
#include <hip/hip_runtime.h>

// Segment-packed (block-diagonal) flash attention for MI455X (gfx1250).
// Pre-pass converts Q(K,V) f32 -> bf16 once (Q pre-scaled by 1/sqrt(d)); main
// kernel: one wave32 per (32-row query tile, head), 16 bf16 WMMAs per 32-key
// tile, K/V tiles staged to LDS with gfx1250 async-to-LDS loads (double
// buffered, ASYNCcnt), f32 online softmax with v_permlane16 reductions.

#define SH 16   // heads
#define SD 64   // head dim

typedef __attribute__((ext_vector_type(16))) __bf16        v16bf;
typedef __attribute__((ext_vector_type(8)))  float         v8f;
typedef __attribute__((ext_vector_type(8)))  unsigned int  v8u;
typedef __attribute__((ext_vector_type(4)))  int           v4i;

#if __has_builtin(__builtin_amdgcn_global_load_async_to_lds_b128)
#define USE_ASYNC 1
#else
#define USE_ASYNC 0
#endif

// async-to-LDS builtin takes (int4 __device__*, int4 __shared__*, Ii, Ii)
#define AS1(p) ((__attribute__((address_space(1))) v4i*)(p))
#define AS3(p) ((__attribute__((address_space(3))) v4i*)(p))

static __device__ __forceinline__ unsigned short f2bf(float f) {
    unsigned int u = __builtin_bit_cast(unsigned int, f);
    u += 0x7FFFu + ((u >> 16) & 1u);      // round-to-nearest-even
    return (unsigned short)(u >> 16);
}
static __device__ __forceinline__ unsigned int pack2(float lo, float hi) {
    return (unsigned int)f2bf(lo) | ((unsigned int)f2bf(hi) << 16);
}
static __device__ __forceinline__ v8f wmma_bf16(v16bf a, v16bf b, v8f c) {
    return __builtin_amdgcn_wmma_f32_16x16x32_bf16(false, a, false, b,
                                                   (short)0, c, false, false);
}
static __device__ __forceinline__ void wait_async32() {
#if USE_ASYNC
#if __has_builtin(__builtin_amdgcn_s_wait_asynccnt)
    __builtin_amdgcn_s_wait_asynccnt(32);
#else
    asm volatile("s_wait_asynccnt 32" ::: "memory");
#endif
    asm volatile("" ::: "memory");
#endif
}
static __device__ __forceinline__ void wait_async0() {
#if USE_ASYNC
#if __has_builtin(__builtin_amdgcn_s_wait_asynccnt)
    __builtin_amdgcn_s_wait_asynccnt(0);
#else
    asm volatile("s_wait_asynccnt 0" ::: "memory");
#endif
    asm volatile("" ::: "memory");
#endif
}

// ---- 16-lane (row-local) xor butterflies via v_permlane16_b32 (VALU, no DS) ----
// Selector nibble pairs for lane^1, ^2, ^4, ^8 inside each 16-lane row.
#define PL_X1_LO 0x67452301u
#define PL_X1_HI 0xEFCDAB89u
#define PL_X2_LO 0x54761032u
#define PL_X2_HI 0xDCFE98BAu
#define PL_X4_LO 0x32107654u
#define PL_X4_HI 0xBA98FEDCu
#define PL_X8_LO 0xFEDCBA98u
#define PL_X8_HI 0x76543210u

static __device__ __forceinline__ float plane16(float x, unsigned sl, unsigned sh) {
#if __has_builtin(__builtin_amdgcn_permlane16)
    unsigned u = __builtin_bit_cast(unsigned int, x);
    u = (unsigned)__builtin_amdgcn_permlane16((int)u, (int)u, (int)sl, (int)sh, false, false);
    return __builtin_bit_cast(float, u);
#else
    int d = (sl == PL_X1_LO) ? 1 : (sl == PL_X2_LO) ? 2 : (sl == PL_X4_LO) ? 4 : 8;
    return __shfl_xor(x, d, 32);
#endif
}
static __device__ __forceinline__ float redmax16(float x) {
    x = fmaxf(x, plane16(x, PL_X1_LO, PL_X1_HI));
    x = fmaxf(x, plane16(x, PL_X2_LO, PL_X2_HI));
    x = fmaxf(x, plane16(x, PL_X4_LO, PL_X4_HI));
    x = fmaxf(x, plane16(x, PL_X8_LO, PL_X8_HI));
    return x;
}
static __device__ __forceinline__ float redsum16(float x) {
    x += plane16(x, PL_X1_LO, PL_X1_HI);
    x += plane16(x, PL_X2_LO, PL_X2_HI);
    x += plane16(x, PL_X4_LO, PL_X4_HI);
    x += plane16(x, PL_X8_LO, PL_X8_HI);
    return x;
}

// contiguous bf16 pair -> packed uint (T=ushort: direct; T=float: convert)
template <typename T> static __device__ __forceinline__ unsigned int ldpair(const T* p);
template <> __device__ __forceinline__ unsigned int ldpair<unsigned short>(const unsigned short* p) {
    return *(const unsigned int*)p;
}
template <> __device__ __forceinline__ unsigned int ldpair<float>(const float* p) {
    return pack2(p[0], p[1]);
}
// strided pair (V fragments: elements SD apart)
template <typename T> static __device__ __forceinline__ unsigned int ldpair_s(const T* p, int st);
template <> __device__ __forceinline__ unsigned int ldpair_s<unsigned short>(const unsigned short* p, int st) {
    return (unsigned int)p[0] | ((unsigned int)p[st] << 16);
}
template <> __device__ __forceinline__ unsigned int ldpair_s<float>(const float* p, int st) {
    return pack2(p[0], p[st]);
}
// Q pair (float path applies 1/sqrt(d); bf16 path is pre-scaled by the pre-pass)
template <typename T> static __device__ __forceinline__ unsigned int ldpair_q(const T* p);
template <> __device__ __forceinline__ unsigned int ldpair_q<unsigned short>(const unsigned short* p) {
    return *(const unsigned int*)p;
}
template <> __device__ __forceinline__ unsigned int ldpair_q<float>(const float* p) {
    return pack2(p[0] * 0.125f, p[1] * 0.125f);
}

// ---------------- pre-pass: f32 -> bf16 (Q scaled) ----------------
__global__ __launch_bounds__(256)
void cvt_bf16_kernel(const float4* __restrict__ q, const float4* __restrict__ k,
                     const float4* __restrict__ v, uint2* __restrict__ qo,
                     uint2* __restrict__ ko, uint2* __restrict__ vo, int n4) {
    int i = blockIdx.x * blockDim.x + threadIdx.x;
    if (i >= n4) return;
    const float s = 0.125f;  // 1/sqrt(64)
    float4 a = q[i];
    qo[i] = make_uint2(pack2(a.x * s, a.y * s), pack2(a.z * s, a.w * s));
    float4 b = k[i];
    ko[i] = make_uint2(pack2(b.x, b.y), pack2(b.z, b.w));
    float4 c = v[i];
    vo[i] = make_uint2(pack2(c.x, c.y), pack2(c.z, c.w));
}

// ---------------- main flash-attention kernel ----------------
template <typename T>
__global__ __launch_bounds__(32)
void fa_seg_kernel(const T* __restrict__ qg, const T* __restrict__ kg,
                   const T* __restrict__ vg, const int* __restrict__ cu,
                   int ncu, float* __restrict__ out, int s) {
    const int lane = threadIdx.x & 31;
    const int n16  = lane & 15;
    const int hi   = lane >> 4;
    const int qbase = blockIdx.x * 32;   // 32 query rows per wave (2 groups of 16)
    const int head  = blockIdx.y;

    constexpr int RB = SD * (int)sizeof(T);       // bytes per row
    constexpr int NCH = RB / 16;                  // 16-byte chunks per row
    __shared__ __align__(16) char kraw[2][32 * RB];
    __shared__ __align__(16) char vraw[2][32 * RB];
    __shared__ __align__(16) unsigned short ptile[2][16 * 32];

    // Per-row segment bounds: group g, slot r -> row qbase + 16g + r + 8*hi.
    int rs[2][8], re[2][8];
#pragma unroll
    for (int g = 0; g < 2; ++g) {
#pragma unroll
        for (int r = 0; r < 8; ++r) {
            int row = qbase + 16 * g + r + 8 * hi;
            int s0 = 0, e0 = s;
            for (int i = 0; i + 1 < ncu; ++i) {
                int a = cu[i], b = cu[i + 1];
                if (a <= row && row < b) { s0 = a; e0 = b; }
            }
            rs[g][r] = s0; re[g][r] = e0;
        }
    }
    int lastrow = qbase + 31 < s ? qbase + 31 : s - 1;
    int kstart = 0, kend = s;
    for (int i = 0; i + 1 < ncu; ++i) {
        int a = cu[i], b = cu[i + 1];
        if (a <= qbase && qbase < b) kstart = a;
        if (a <= lastrow && lastrow < b) kend = b;
    }
    kstart &= ~31;

    // Q A-fragments: [group][head-dim chunk]
    v16bf qa[2][2];
#pragma unroll
    for (int g = 0; g < 2; ++g) {
        int qr = qbase + 16 * g + n16;
        if (qr >= s) qr = s - 1;
        const T* qrow = qg + ((size_t)qr * SH + head) * SD;
#pragma unroll
        for (int c = 0; c < 2; ++c) {
            v8u au;
#pragma unroll
            for (int j = 0; j < 8; ++j) {
                int kk = 2 * (j & 3) + 8 * hi + 16 * (j >> 2) + 32 * c;
                au[j] = ldpair_q(&qrow[kk]);
            }
            qa[g][c] = __builtin_bit_cast(v16bf, au);
        }
    }

    v8f o[2][4];
    float mrow[2][8], lrow[2][8];
#pragma unroll
    for (int g = 0; g < 2; ++g) {
#pragma unroll
        for (int c = 0; c < 4; ++c) o[g][c] = v8f{};
#pragma unroll
        for (int r = 0; r < 8; ++r) { mrow[g][r] = -3.0e38f; lrow[g][r] = 0.0f; }
    }

    // Stage one 32-key K/V tile into LDS buffer b (async on gfx1250).
    auto stage_tile = [&](int ktb, int b) {
#pragma unroll
        for (int i = 0; i < NCH; ++i) {
            int cid = i * 32 + lane;
            int row = cid / NCH;
            int col = cid % NCH;
            int key = ktb + row;
            if (key >= s) key = s - 1;
            const char* gk = (const char*)kg + ((size_t)key * SH + head) * RB + col * 16;
            const char* gv = (const char*)vg + ((size_t)key * SH + head) * RB + col * 16;
            char* lk = &kraw[b][row * RB + col * 16];
            char* lv = &vraw[b][row * RB + col * 16];
#if USE_ASYNC
            __builtin_amdgcn_global_load_async_to_lds_b128(AS1(gk), AS3(lk), 0, 0);
            __builtin_amdgcn_global_load_async_to_lds_b128(AS1(gv), AS3(lv), 0, 0);
#else
            *(uint4*)lk = *(const uint4*)gk;
            *(uint4*)lv = *(const uint4*)gv;
#endif
        }
    };

    int ntiles = (kend - kstart + 31) / 32;
    stage_tile(kstart, 0);

    for (int t = 0; t < ntiles; ++t) {
        int kt  = kstart + t * 32;
        int buf = t & 1;
        bool more = (t + 1) < ntiles;
        if (more) stage_tile(kstart + (t + 1) * 32, buf ^ 1);
#if USE_ASYNC
        if (more) wait_async32(); else wait_async0();   // tile t complete (in-order)
#else
        __syncthreads();
#endif
        const T* kb = (const T*)kraw[buf];
        const T* vb = (const T*)vraw[buf];

        // ---- S = Q K^T : per group, two 16x16 f32 fragments ----
        v8f c0g[2], c1g[2];
#pragma unroll
        for (int g = 0; g < 2; ++g) { c0g[g] = v8f{}; c1g[g] = v8f{}; }
#pragma unroll
        for (int c = 0; c < 2; ++c) {
            v8u b0u, b1u;
#pragma unroll
            for (int j = 0; j < 8; ++j) {
                int f = 2 * (j & 3) + 8 * hi + 16 * (j >> 2) + 32 * c;
                b0u[j] = ldpair(&kb[n16 * SD + f]);          // keys 0-15
                b1u[j] = ldpair(&kb[(n16 + 16) * SD + f]);   // keys 16-31
            }
            v16bf b0 = __builtin_bit_cast(v16bf, b0u);
            v16bf b1 = __builtin_bit_cast(v16bf, b1u);
#pragma unroll
            for (int g = 0; g < 2; ++g) {
                c0g[g] = wmma_bf16(qa[g][c], b0, c0g[g]);
                c1g[g] = wmma_bf16(qa[g][c], b1, c1g[g]);
            }
        }

        // ---- online softmax per group (permlane16 row reductions) ----
        int kc0 = kt + n16, kc1 = kt + n16 + 16;
#pragma unroll
        for (int g = 0; g < 2; ++g) {
#pragma unroll
            for (int r = 0; r < 8; ++r) {
                bool v0 = (kc0 >= rs[g][r]) && (kc0 < re[g][r]);
                bool v1 = (kc1 >= rs[g][r]) && (kc1 < re[g][r]);
                float t0 = fmaxf(v0 ? c0g[g][r] : -3.0e38f, v1 ? c1g[g][r] : -3.0e38f);
                t0 = redmax16(t0);
                float mnew = fmaxf(mrow[g][r], t0);
                float p0 = v0 ? __expf(c0g[g][r] - mnew) : 0.0f;
                float p1 = v1 ? __expf(c1g[g][r] - mnew) : 0.0f;
                float rsum = redsum16(p0 + p1);
                float alpha = __expf(mrow[g][r] - mnew);
                lrow[g][r] = lrow[g][r] * alpha + rsum;
                mrow[g][r] = mnew;
#pragma unroll
                for (int c = 0; c < 4; ++c) o[g][c][r] *= alpha;
                int m = r + 8 * hi;
                ptile[g][m * 32 + n16]      = f2bf(p0);
                ptile[g][m * 32 + n16 + 16] = f2bf(p1);
            }
        }
        __syncthreads();

        // ---- O += P (16x32) x V (32x64) per group ----
        v16bf pa[2];
#pragma unroll
        for (int g = 0; g < 2; ++g) {
            v8u pu;
#pragma unroll
            for (int j = 0; j < 8; ++j) {
                int kk = 2 * (j & 3) + 8 * hi + 16 * (j >> 2);
                pu[j] = *(const unsigned int*)&ptile[g][n16 * 32 + kk];
            }
            pa[g] = __builtin_bit_cast(v16bf, pu);
        }
#pragma unroll
        for (int c = 0; c < 4; ++c) {
            v8u bu;
#pragma unroll
            for (int j = 0; j < 8; ++j) {
                int kk = 2 * (j & 3) + 8 * hi + 16 * (j >> 2);
                bu[j] = ldpair_s(&vb[kk * SD + 16 * c + n16], SD);
            }
            v16bf bv = __builtin_bit_cast(v16bf, bu);
#pragma unroll
            for (int g = 0; g < 2; ++g)
                o[g][c] = wmma_bf16(pa[g], bv, o[g][c]);
        }
        __syncthreads();
    }

    // ---- epilogue: normalize, store f32 ----
#pragma unroll
    for (int g = 0; g < 2; ++g) {
#pragma unroll
        for (int c = 0; c < 4; ++c) {
#pragma unroll
            for (int r = 0; r < 8; ++r) {
                int row = qbase + 16 * g + r + 8 * hi;
                if (row < s) {
                    float inv = (lrow[g][r] > 0.0f) ? (1.0f / lrow[g][r]) : 0.0f;
                    out[((size_t)row * SH + head) * SD + 16 * c + n16] = o[g][c][r] * inv;
                }
            }
        }
    }
}

extern "C" void kernel_launch(void* const* d_in, const int* in_sizes, int n_in,
                              void* d_out, int out_size, void* d_ws, size_t ws_size,
                              hipStream_t stream) {
    const float* q  = (const float*)d_in[0];
    const float* k  = (const float*)d_in[1];
    const float* v  = (const float*)d_in[2];
    // d_in[3] = bsz (==1); layout is already [b*s, h, d]
    const int*   cu = (const int*)d_in[4];
    int ncu   = in_sizes[4];
    int total = in_sizes[0];          // S*H*D
    int s     = total / (SH * SD);
    float* out = (float*)d_out;

    dim3 grid((s + 31) / 32, SH);
    dim3 block(32);

    size_t need = (size_t)total * sizeof(unsigned short) * 3;
    if (ws_size >= need) {
        unsigned short* qb = (unsigned short*)d_ws;
        unsigned short* kb = qb + total;
        unsigned short* vb = kb + total;
        int n4 = total / 4;
        cvt_bf16_kernel<<<(n4 + 255) / 256, 256, 0, stream>>>(
            (const float4*)q, (const float4*)k, (const float4*)v,
            (uint2*)qb, (uint2*)kb, (uint2*)vb, n4);
        fa_seg_kernel<unsigned short><<<grid, block, 0, stream>>>(qb, kb, vb, cu, ncu, out, s);
    } else {
        fa_seg_kernel<float><<<grid, block, 0, stream>>>(q, k, v, cu, ncu, out, s);
    }
}
